// SpaceSelfAttentionModality_67138928771076
// MI455X (gfx1250) — matrix operand order repaired and checked by reading, not compiled, not run on hardware
//
#include <hip/hip_runtime.h>
#include <stdint.h>

// Problem constants
#define NROWS 16384          // B*T*S = 4*8*512
#define Dv    1024
#define Sv    512
#define Hv    16
#define HDv   64
#define KDW   512            // dwords per row of D (bf16 pairs)

typedef __attribute__((ext_vector_type(16))) __bf16 bf16x16;
typedef __attribute__((ext_vector_type(8)))  float  f32x8;

union BFrag { uint32_t u[8]; uint4 q[2]; bf16x16 v; };

__device__ __forceinline__ f32x8 zero8() {
  f32x8 z = {0.f,0.f,0.f,0.f,0.f,0.f,0.f,0.f};
  return z;
}

__device__ __forceinline__ uint16_t f2bf(float f) {
  uint32_t u = __builtin_bit_cast(uint32_t, f);
  u += 0x7fffu + ((u >> 16) & 1u);     // round-to-nearest-even
  return (uint16_t)(u >> 16);
}

// ---- CDNA5 async global->LDS copy (16B per lane), tracked by ASYNCcnt ----
__device__ __forceinline__ void async_copy16(uint32_t lds_off, const void* gptr) {
  asm volatile("global_load_async_to_lds_b128 %0, %1, off"
               :: "v"(lds_off), "v"(gptr)
               : "memory");
}
__device__ __forceinline__ void wait_async0() { asm volatile("s_wait_asynccnt 0x0" ::: "memory"); }
__device__ __forceinline__ void wait_async1() { asm volatile("s_wait_asynccnt 0x1" ::: "memory"); }
__device__ __forceinline__ void wait_async2() { asm volatile("s_wait_asynccnt 0x2" ::: "memory"); }

// Fragment layouts used here (contiguous k, same for A and B; not the ISA table order), expressed as two 16B groups per lane:
// A (16x32 bf16): lane m=lane&15, half=lane>>4:
//   u[0..7] = dwords [half*8 .. half*8+7]   (two contiguous uint4)
// B (32x16 bf16): lane n=lane&15, half:
//   u[0..7] = dwords [half*8 .. half*8+7]   (two contiguous uint4)

__device__ __forceinline__ BFrag load_afrag_g(const uint32_t* __restrict__ base,
                                              int row0, int row_dw, int chunk_dw) {
  const int lane = threadIdx.x & 31, half = lane >> 4, m = lane & 15;
  const uint32_t* p = base + (size_t)(row0 + m) * row_dw + chunk_dw + half * 8;
  BFrag f;
  f.q[0] = *(const uint4*)(p);
  f.q[1] = *(const uint4*)(p + 4);
  return f;
}
__device__ __forceinline__ BFrag load_bfrag_g(const uint32_t* __restrict__ base,
                                              int col0, int row_dw, int chunk_dw) {
  const int lane = threadIdx.x & 31, half = lane >> 4, n = lane & 15;
  const uint32_t* p = base + (size_t)(col0 + n) * row_dw + chunk_dw + half * 8;
  BFrag f;
  f.q[0] = *(const uint4*)(p);
  f.q[1] = *(const uint4*)(p + 4);
  return f;
}
// rowbase must be 16B aligned (multiple of 16 dwords from an aligned array)
__device__ __forceinline__ BFrag load_bfrag_lds(const uint32_t* rowbase, int half) {
  const uint32_t* p = rowbase + half * 8;
  BFrag f;
  f.q[0] = *(const uint4*)(p);
  f.q[1] = *(const uint4*)(p + 4);
  return f;
}
__device__ __forceinline__ BFrag load_afrag_lds(const uint32_t* rowbase, int half) {
  const uint32_t* p = rowbase + half * 8;
  BFrag f;
  f.q[0] = *(const uint4*)(p);
  f.q[1] = *(const uint4*)(p + 4);
  return f;
}

__device__ __forceinline__ f32x8 wmma_bf16(const BFrag& a, const BFrag& b, f32x8 c) {
  return __builtin_amdgcn_wmma_f32_16x16x32_bf16(false, a.v, false, b.v, (short)0, c, false, false);
}

// ---------------- converters ----------------
__global__ void cvt_bf16_kernel(const float* __restrict__ src, uint16_t* __restrict__ dst, int n) {
  int i = blockIdx.x * blockDim.x + threadIdx.x;
  int stride = gridDim.x * blockDim.x;
  for (; i < n; i += stride) dst[i] = f2bf(src[i]);
}

__global__ void cvt_transpose_kernel(const float* __restrict__ W, uint16_t* __restrict__ wT,
                                     int R, int C) {
  int i = blockIdx.x * blockDim.x + threadIdx.x;
  int n = R * C;
  int stride = gridDim.x * blockDim.x;
  for (; i < n; i += stride) {
    int c = i / R, r = i - c * R;
    wT[(size_t)c * R + r] = f2bf(W[(size_t)r * C + c]);
  }
}

// ---------------- QKV GEMM + bias + RMSNorm epilogue ----------------
// grid.x = column block (48: 16 q-heads, 16 k-heads, 16 v-heads), grid.y = row block (64)
// q,k written [n,h,s,hd]; v written TRANSPOSED [n,h,hd,s] for async V-tile staging.
__global__ __launch_bounds__(256)
void qkv_gemm_kernel(const uint32_t* __restrict__ xb, const uint32_t* __restrict__ wT,
                     const float* __restrict__ bqkv,
                     const float* __restrict__ qscale, const float* __restrict__ kscale,
                     uint16_t* __restrict__ qw, uint16_t* __restrict__ kw,
                     uint16_t* __restrict__ vw) {
  __shared__ __align__(16) uint32_t sB[2][64 * 16];  // double-buffered B tile
  __shared__ float ylds[256][64];
  __shared__ float sbias[64];
  __shared__ float sscl[64];

  const int tid = threadIdx.x;
  const int wave = tid >> 5, lane = tid & 31, half = lane >> 4;
  const int cb = blockIdx.x;
  const int rowblk = blockIdx.y * 256;
  const int kind = cb >> 4;             // 0=q 1=k 2=v
  const int head = cb & 15;

  if (tid < 64) {
    sbias[tid] = bqkv[cb * 64 + tid];
    sscl[tid] = (kind == 0) ? qscale[tid] : (kind == 1 ? kscale[tid] : 1.0f);
  }

  f32x8 cc[2][4];
#pragma unroll
  for (int rs = 0; rs < 2; ++rs)
#pragma unroll
    for (int t = 0; t < 4; ++t) cc[rs][t] = zero8();

  const int arow = rowblk + wave * 32;

  // each thread owns one 16B slot of the B tile
  const int cslot = tid >> 2, islot = (tid & 3) * 4;
  const uint32_t* gsrc = wT + (size_t)(cb * 64 + cslot) * KDW + islot;
  uint32_t ldcur = (uint32_t)(uintptr_t)&sB[0][cslot * 16 + islot];
  uint32_t ldnxt = (uint32_t)(uintptr_t)&sB[1][cslot * 16 + islot];
  const uint32_t* sbcur = &sB[0][0];
  const uint32_t* sbnxt = &sB[1][0];

  auto do_chunk = [&](int cdw, const uint32_t* sb) {
    BFrag a0 = load_afrag_g(xb, arow, KDW, cdw);
    BFrag a1 = load_afrag_g(xb, arow + 16, KDW, cdw);
#pragma unroll
    for (int t = 0; t < 4; ++t) {
      BFrag bb = load_bfrag_lds(sb + (t * 16 + (lane & 15)) * 16, half);
      cc[0][t] = wmma_bf16(a0, bb, cc[0][t]);
      cc[1][t] = wmma_bf16(a1, bb, cc[1][t]);
    }
  };

  async_copy16(ldcur, gsrc);            // prefetch chunk 0

#pragma unroll 1
  for (int chunk = 0; chunk < 31; ++chunk) {
    __syncthreads();                    // readers of the other buffer are done
    async_copy16(ldnxt, gsrc + (chunk + 1) * 16);
    wait_async1();                      // my chunk's copy arrived; next in flight
    __syncthreads();                    // everyone's chunk data in LDS
    do_chunk(chunk * 16, sbcur);
    uint32_t tl = ldcur; ldcur = ldnxt; ldnxt = tl;
    const uint32_t* tp = sbcur; sbcur = sbnxt; sbnxt = tp;
  }
  __syncthreads();
  wait_async0();
  __syncthreads();
  do_chunk(31 * 16, sbcur);             // peeled final chunk

  __syncthreads();
#pragma unroll
  for (int rs = 0; rs < 2; ++rs)
#pragma unroll
    for (int t = 0; t < 4; ++t)
#pragma unroll
      for (int r = 0; r < 8; ++r)
        ylds[wave * 32 + rs * 16 + r + half * 8][t * 16 + (lane & 15)] = cc[rs][t][r];
  __syncthreads();

  {  // per-row epilogue: thread tid owns block row tid
    const int grow = rowblk + tid;
    const int n = grow >> 9, s = grow & 511;
    if (kind < 2) {
      float ms = 0.f;
#pragma unroll 8
      for (int j = 0; j < 64; ++j) {
        float v = ylds[tid][j] + sbias[j];
        ms += v * v;
      }
      ms *= (1.0f / 64.0f);
      const float rinv = rsqrtf(ms + 1e-6f);
      uint16_t* dst = (kind == 0 ? qw : kw);
      uint32_t* d32 = (uint32_t*)(dst + ((size_t)(n * Hv + head) * Sv + s) * HDv);
#pragma unroll 8
      for (int j = 0; j < 64; j += 2) {
        float v0 = (ylds[tid][j] + sbias[j]) * sscl[j] * rinv;
        float v1 = (ylds[tid][j + 1] + sbias[j + 1]) * sscl[j + 1] * rinv;
        d32[j >> 1] = (uint32_t)f2bf(v0) | ((uint32_t)f2bf(v1) << 16);
      }
    } else {  // V: store transposed [n,h,hd,s]
      uint16_t* dstv = vw + (size_t)(n * Hv + head) * HDv * Sv;
#pragma unroll 8
      for (int j = 0; j < 64; ++j) {
        float v0 = ylds[tid][j] + sbias[j];
        dstv[(size_t)j * Sv + s] = f2bf(v0);
      }
    }
  }
}

// ---------------- flash attention ----------------
// grid.x = query block (4 x 128q), grid.y = n*16+h (512). 8 waves x 16 queries.
// vw is [n,h,hd,s] so V tiles stage via async b128 copies, already transposed.
__global__ __launch_bounds__(256)
void attn_kernel(const uint16_t* __restrict__ qw, const uint16_t* __restrict__ kw,
                 const uint16_t* __restrict__ vw, const int* __restrict__ mod,
                 uint16_t* __restrict__ yb) {
  __shared__ __align__(16) uint16_t vT[2][64][64];     // [buf][hd][key_local]
  __shared__ __align__(16) uint16_t pbuf[8][16][64];   // per-wave P tile (bf16)
  __shared__ int smod[Sv];

  const int tid = threadIdx.x;
  const int wave = tid >> 5, lane = tid & 31, half = lane >> 4;
  const int nh = blockIdx.y;
  const int n = nh >> 4, h = nh & 15;
  const int qbase = blockIdx.x * 128 + wave * 16;

  for (int i = tid; i < Sv; i += 256) smod[i] = mod[i];

  const uint32_t* q32 = (const uint32_t*)qw + (size_t)nh * Sv * (HDv / 2);
  const uint32_t* k32 = (const uint32_t*)kw + (size_t)nh * Sv * (HDv / 2);
  const uint16_t* vtp = vw + (size_t)nh * HDv * Sv;    // [hd][s]

  BFrag qa0 = load_afrag_g(q32, qbase, HDv / 2, 0);
  BFrag qa1 = load_afrag_g(q32, qbase, HDv / 2, 16);

  // per-thread V-copy slot: 64 hd-rows x 128B = 512 x 16B segments, 2 per thread
  const int hd0 = tid >> 3,          seg0 = tid & 7;
  const int hd1 = (tid + 256) >> 3,  seg1 = tid & 7;
  uint32_t vld0c = (uint32_t)(uintptr_t)&vT[0][hd0][seg0 * 8];
  uint32_t vld1c = (uint32_t)(uintptr_t)&vT[0][hd1][seg1 * 8];
  uint32_t vld0n = (uint32_t)(uintptr_t)&vT[1][hd0][seg0 * 8];
  uint32_t vld1n = (uint32_t)(uintptr_t)&vT[1][hd1][seg1 * 8];
  const uint32_t* vdwcur = (const uint32_t*)&vT[0][0][0];
  const uint32_t* vdwnxt = (const uint32_t*)&vT[1][0][0];

  f32x8 acc[4];
#pragma unroll
  for (int t = 0; t < 4; ++t) acc[t] = zero8();
  float m[8], l[8];
  int qmods[8]; bool qlat[8];
  __syncthreads();
#pragma unroll
  for (int r = 0; r < 8; ++r) {
    m[r] = -3.0e38f; l[r] = 0.f;
    int qa = qbase + r + half * 8;
    qlat[r] = qa < 64;                  // latents attend everywhere
    qmods[r] = smod[qa];
  }

  auto do_kblock = [&](int kb0, const uint32_t* vdw) {
    // scores S = (q @ k^T) * 1/8, masked
    f32x8 st[4];
#pragma unroll
    for (int t = 0; t < 4; ++t) {
      BFrag b0 = load_bfrag_g(k32, kb0 + t * 16, HDv / 2, 0);
      BFrag b1 = load_bfrag_g(k32, kb0 + t * 16, HDv / 2, 16);
      f32x8 s = zero8();
      s = wmma_bf16(qa0, b0, s);
      s = wmma_bf16(qa1, b1, s);
      const int kmod = smod[kb0 + t * 16 + (lane & 15)];
#pragma unroll
      for (int r = 0; r < 8; ++r) {
        bool allow = qlat[r] || (qmods[r] == kmod);
        st[t][r] = allow ? s[r] * 0.125f : -3.0e38f;
      }
    }

    // online softmax, rows reduced across 16-lane halves
#pragma unroll
    for (int r = 0; r < 8; ++r) {
      float mx = fmaxf(fmaxf(st[0][r], st[1][r]), fmaxf(st[2][r], st[3][r]));
#pragma unroll
      for (int o = 1; o < 16; o <<= 1) mx = fmaxf(mx, __shfl_xor(mx, o, 32));
      float mnew = fmaxf(m[r], mx);
      float alpha = __expf(m[r] - mnew);
      m[r] = mnew;
      float ssum = 0.f;
#pragma unroll
      for (int t = 0; t < 4; ++t) {
        float p = __expf(st[t][r] - mnew);
        ssum += p;
        pbuf[wave][r + half * 8][t * 16 + (lane & 15)] = f2bf(p);
      }
#pragma unroll
      for (int o = 1; o < 16; o <<= 1) ssum += __shfl_xor(ssum, o, 32);
      l[r] = l[r] * alpha + ssum;
#pragma unroll
      for (int t = 0; t < 4; ++t) acc[t][r] = acc[t][r] * alpha;
    }
    __syncthreads();

    // acc += P @ V
    const uint32_t* pdw = (const uint32_t*)&pbuf[wave][0][0];
    BFrag pa0 = load_afrag_lds(pdw + (lane & 15) * 32, half);
    BFrag pa1 = load_afrag_lds(pdw + (lane & 15) * 32 + 16, half);
#pragma unroll
    for (int t = 0; t < 4; ++t) {
      BFrag vb0 = load_bfrag_lds(vdw + (t * 16 + (lane & 15)) * 32, half);
      BFrag vb1 = load_bfrag_lds(vdw + (t * 16 + (lane & 15)) * 32 + 16, half);
      acc[t] = wmma_bf16(pa0, vb0, acc[t]);
      acc[t] = wmma_bf16(pa1, vb1, acc[t]);
    }
  };

  // prefetch key-block 0
  async_copy16(vld0c, vtp + (size_t)hd0 * Sv + seg0 * 8);
  async_copy16(vld1c, vtp + (size_t)hd1 * Sv + seg1 * 8);

#pragma unroll 1
  for (int kb = 0; kb < 7; ++kb) {
    const int kb0 = kb * 64;
    __syncthreads();                    // prior P@V reads of other buffer done
    async_copy16(vld0n, vtp + (size_t)hd0 * Sv + kb0 + 64 + seg0 * 8);
    async_copy16(vld1n, vtp + (size_t)hd1 * Sv + kb0 + 64 + seg1 * 8);
    wait_async2();                      // this block's 2 copies arrived
    __syncthreads();
    do_kblock(kb0, vdwcur);
    uint32_t t0 = vld0c; vld0c = vld0n; vld0n = t0;
    uint32_t t1 = vld1c; vld1c = vld1n; vld1n = t1;
    const uint32_t* tp = vdwcur; vdwcur = vdwnxt; vdwnxt = tp;
  }
  __syncthreads();
  wait_async0();
  __syncthreads();
  do_kblock(7 * 64, vdwcur);            // peeled final key block

  // y = acc / l, back to [n, s, h*64+hd] bf16
#pragma unroll
  for (int t = 0; t < 4; ++t)
#pragma unroll
    for (int r = 0; r < 8; ++r) {
      int qabs = qbase + r + half * 8;
      float yv = acc[t][r] / l[r];
      yb[((size_t)(n * Sv + qabs)) * Dv + h * HDv + t * 16 + (lane & 15)] = f2bf(yv);
    }
}

// ---------------- output projection ----------------
__global__ __launch_bounds__(256)
void out_gemm_kernel(const uint32_t* __restrict__ ybd, const uint32_t* __restrict__ woT,
                     const float* __restrict__ bout, float* __restrict__ out) {
  __shared__ __align__(16) uint32_t sB[2][64 * 16];
  const int tid = threadIdx.x;
  const int wave = tid >> 5, lane = tid & 31, half = lane >> 4;
  const int cb = blockIdx.x;            // 0..15
  const int rowblk = blockIdx.y * 256;

  f32x8 cc[2][4];
#pragma unroll
  for (int rs = 0; rs < 2; ++rs)
#pragma unroll
    for (int t = 0; t < 4; ++t) cc[rs][t] = zero8();

  const int arow = rowblk + wave * 32;

  const int cslot = tid >> 2, islot = (tid & 3) * 4;
  const uint32_t* gsrc = woT + (size_t)(cb * 64 + cslot) * KDW + islot;
  uint32_t ldcur = (uint32_t)(uintptr_t)&sB[0][cslot * 16 + islot];
  uint32_t ldnxt = (uint32_t)(uintptr_t)&sB[1][cslot * 16 + islot];
  const uint32_t* sbcur = &sB[0][0];
  const uint32_t* sbnxt = &sB[1][0];

  auto do_chunk = [&](int cdw, const uint32_t* sb) {
    BFrag a0 = load_afrag_g(ybd, arow, KDW, cdw);
    BFrag a1 = load_afrag_g(ybd, arow + 16, KDW, cdw);
#pragma unroll
    for (int t = 0; t < 4; ++t) {
      BFrag bb = load_bfrag_lds(sb + (t * 16 + (lane & 15)) * 16, half);
      cc[0][t] = wmma_bf16(a0, bb, cc[0][t]);
      cc[1][t] = wmma_bf16(a1, bb, cc[1][t]);
    }
  };

  async_copy16(ldcur, gsrc);

#pragma unroll 1
  for (int chunk = 0; chunk < 31; ++chunk) {
    __syncthreads();
    async_copy16(ldnxt, gsrc + (chunk + 1) * 16);
    wait_async1();
    __syncthreads();
    do_chunk(chunk * 16, sbcur);
    uint32_t tl = ldcur; ldcur = ldnxt; ldnxt = tl;
    const uint32_t* tp = sbcur; sbcur = sbnxt; sbnxt = tp;
  }
  __syncthreads();
  wait_async0();
  __syncthreads();
  do_chunk(31 * 16, sbcur);

#pragma unroll
  for (int rs = 0; rs < 2; ++rs)
#pragma unroll
    for (int t = 0; t < 4; ++t) {
      int col = cb * 64 + t * 16 + (lane & 15);
      float bias = bout[col];
#pragma unroll
      for (int r = 0; r < 8; ++r) {
        int grow = arow + rs * 16 + r + half * 8;
        out[(size_t)grow * Dv + col] = cc[rs][t][r] + bias;
      }
    }
}

// ---------------- launch ----------------
extern "C" void kernel_launch(void* const* d_in, const int* in_sizes, int n_in,
                              void* d_out, int out_size, void* d_ws, size_t ws_size,
                              hipStream_t stream) {
  const float* x      = (const float*)d_in[0];
  const int*   mids   = (const int*)  d_in[1];
  const float* Wqkv   = (const float*)d_in[2];
  const float* bqkv   = (const float*)d_in[3];
  const float* Wout   = (const float*)d_in[4];
  const float* bout   = (const float*)d_in[5];
  const float* qscale = (const float*)d_in[6];
  const float* kscale = (const float*)d_in[7];
  float* out = (float*)d_out;

  char* ws = (char*)d_ws;
  size_t off = 0;
  auto alloc = [&](size_t bytes) -> char* {
    char* p = ws + off;
    off += (bytes + 255) & ~(size_t)255;
    return p;
  };
  uint16_t* xb    = (uint16_t*)alloc((size_t)NROWS * Dv * 2);          // 33.5 MB
  uint16_t* wqkvT = (uint16_t*)alloc((size_t)3 * Dv * Dv * 2);         // 6.3 MB
  uint16_t* woutT = (uint16_t*)alloc((size_t)Dv * Dv * 2);             // 2.1 MB
  uint16_t* qwb   = (uint16_t*)alloc((size_t)32 * Hv * Sv * HDv * 2);  // 33.5 MB
  uint16_t* kwb   = (uint16_t*)alloc((size_t)32 * Hv * Sv * HDv * 2);
  uint16_t* vwb   = (uint16_t*)alloc((size_t)32 * Hv * Sv * HDv * 2);  // transposed [n,h,hd,s]
  uint16_t* ybw   = (uint16_t*)alloc((size_t)NROWS * Dv * 2);          // 33.5 MB

  cvt_bf16_kernel<<<4096, 256, 0, stream>>>(x, xb, NROWS * Dv);
  cvt_transpose_kernel<<<2048, 256, 0, stream>>>(Wqkv, wqkvT, Dv, 3 * Dv);
  cvt_transpose_kernel<<<1024, 256, 0, stream>>>(Wout, woutT, Dv, Dv);
  qkv_gemm_kernel<<<dim3(48, 64), 256, 0, stream>>>(
      (const uint32_t*)xb, (const uint32_t*)wqkvT, bqkv, qscale, kscale, qwb, kwb, vwb);
  attn_kernel<<<dim3(4, 512), 256, 0, stream>>>(qwb, kwb, vwb, mids, ybw);
  out_gemm_kernel<<<dim3(16, 64), 256, 0, stream>>>(
      (const uint32_t*)ybw, (const uint32_t*)woutT, bout, out);
}
